// GNNLayer_24790551232796
// MI455X (gfx1250) — compile-verified, matrix operand order checked by reading, not run on hardware
//
#include <hip/hip_runtime.h>
#include <hip/hip_bf16.h>
#include <math.h>

typedef __attribute__((ext_vector_type(16))) _Float16 v16h;
typedef __attribute__((ext_vector_type(8)))  _Float16 v8h;
typedef __attribute__((ext_vector_type(4)))  _Float16 v4h;
typedef __attribute__((ext_vector_type(8)))  float    v8f;

#define BB    2
#define NN    4096
#define DIM   128
#define TOPK  4
#define CHUNK 1024   // candidates staged in LDS per block iteration (16 KB)

// sorted-insert into a per-lane ascending top-4 (smallest d2 first).
// tie-break: lower index wins (matches jax.lax.top_k on the adjacency).
__device__ __forceinline__ void insert4(float d, int id, float dd[4], int ii[4]) {
    bool w3 = (d < dd[3]) || (d == dd[3] && id < ii[3]);
    if (!w3) return;
    if ((d < dd[0]) || (d == dd[0] && id < ii[0])) {
        dd[3]=dd[2]; ii[3]=ii[2]; dd[2]=dd[1]; ii[2]=ii[1];
        dd[1]=dd[0]; ii[1]=ii[0]; dd[0]=d;     ii[0]=id;
    } else if ((d < dd[1]) || (d == dd[1] && id < ii[1])) {
        dd[3]=dd[2]; ii[3]=ii[2]; dd[2]=dd[1]; ii[2]=ii[1];
        dd[1]=d;     ii[1]=id;
    } else if ((d < dd[2]) || (d == dd[2] && id < ii[2])) {
        dd[3]=dd[2]; ii[3]=ii[2]; dd[2]=d;     ii[2]=id;
    } else {
        dd[3]=d;     ii[3]=id;
    }
}

// ---------------- Kernel 0: prep ----------------
// t < 16384 : repack W (f32 [K=128][N=128]) into f16 WMMA B-fragment layout
//   Wh[((nt*4+kc)*32+lane)*16+e] = W[(kc*32 + koff)*128 + nt*16 + (lane&15)]
//   koff = (e&7) + ((e>>3)<<4) + (lane>>4)*8   (CDNA5 16-bit B striping)
// else      : pack pos [B*N][3] f32 into float4 (x,y,z,0) for b128 loads
__global__ __launch_bounds__(256) void prep_kernel(const float* __restrict__ W,
                                                   const float* __restrict__ pos,
                                                   _Float16* __restrict__ Wh,
                                                   float4* __restrict__ pos4) {
    int t = blockIdx.x * 256 + threadIdx.x;      // 24576 total
    if (t < 16384) {
        int e    = t & 15;
        int lane = (t >> 4) & 31;
        int kc   = (t >> 9) & 3;
        int nt   = t >> 11;
        int n    = nt * 16 + (lane & 15);
        int half = lane >> 4;
        int koff = (e & 7) + ((e >> 3) << 4) + half * 8;
        Wh[t] = (_Float16)W[(kc * 32 + koff) * DIM + n];
    } else {
        int m = t - 16384;                       // [0, B*N)
        pos4[m] = make_float4(pos[m*3+0], pos[m*3+1], pos[m*3+2], 0.0f);
    }
}

// ---------------- Kernel 1: 4-NN selection ----------------
// 8 waves per block, one node per wave; all 8 waves share pos chunks staged
// in LDS via global_load_async_to_lds_b128 (ASYNCcnt path) -> scan runs on
// ds_load_b128 with 1 compare + 6 VALU per candidate.
__global__ __launch_bounds__(256) void topk_kernel(const float4* __restrict__ pos4,
                                                   int*   __restrict__ idx_out,
                                                   float* __restrict__ w_out) {
    __shared__ float4 pp[CHUNK];                 // 16 KB staging buffer

    const int tid  = threadIdx.x;
    const int wave = tid >> 5;
    const int lane = tid & 31;
    const int row  = blockIdx.x * 8 + wave;      // [0, B*N)
    const int b    = row / NN;
    const int n    = row % NN;
    const float4* pb = pos4 + (size_t)b * NN;

    const float4 q = pb[n];

    float dd[TOPK] = {1e30f, 1e30f, 1e30f, 1e30f};
    int   ii[TOPK] = {0x7fffffff, 0x7fffffff, 0x7fffffff, 0x7fffffff};

    for (int c0 = 0; c0 < NN; c0 += CHUNK) {
        __syncthreads();                         // previous scan done before refill
        // cooperative async fill: 1024 float4 by 256 lanes (4 each), 16B/lane.
        // VDST = LDS byte address (low 32 bits of generic pointer per LDS
        // aperture truncation), VADDR = 64-bit global address pair, SADDR=off.
        #pragma unroll
        for (int i = 0; i < CHUNK / 256; ++i) {
            uint32_t      laddr = (uint32_t)(uintptr_t)(&pp[tid + i * 256]);
            const float4* gaddr = pb + c0 + tid + i * 256;
            asm volatile("global_load_async_to_lds_b128 %0, %1, off"
                         :
                         : "v"(laddr), "v"(gaddr)
                         : "memory");
        }
        if (c0 + CHUNK < NN)                     // warm L2/WGP$ for next chunk
            __builtin_prefetch(pb + c0 + CHUNK + tid, 0, 0);
        asm volatile("s_wait_asynccnt 0x0" ::: "memory");
        __syncthreads();

        // scan chunk from LDS
        #pragma unroll 4
        for (int j = lane; j < CHUNK; j += 32) {
            float4 p = pp[j];
            float d0 = p.x - q.x;
            float d1 = p.y - q.y;
            float d2 = p.z - q.z;
            float d2s = fmaf(d0, d0, fmaf(d1, d1, d2 * d2));
            if (d2s <= dd[3]) insert4(d2s, c0 + j, dd, ii);   // rare path
        }
    }

    // butterfly merge of per-lane top-4 lists across the wave
    #pragma unroll
    for (int off = 16; off >= 1; off >>= 1) {
        float od[TOPK]; int oi[TOPK];
        #pragma unroll
        for (int k = 0; k < TOPK; ++k) {
            od[k] = __shfl_xor(dd[k], off, 32);
            oi[k] = __shfl_xor(ii[k], off, 32);
        }
        #pragma unroll
        for (int k = 0; k < TOPK; ++k) insert4(od[k], oi[k], dd, ii);
    }

    if (lane == 0) {
        // adj = exp(-dist^2/2), dist = sqrt(d2+1e-8)  =>  adj = exp(-(d2+1e-8)/2)
        float a[TOPK], s = 0.0f;
        #pragma unroll
        for (int k = 0; k < TOPK; ++k) { a[k] = __expf(-0.5f * (dd[k] + 1e-8f)); s += a[k]; }
        float inv = 1.0f / (s + 1e-8f);
        #pragma unroll
        for (int k = 0; k < TOPK; ++k) {
            idx_out[row*TOPK + k] = ii[k];
            w_out [row*TOPK + k] = a[k] * inv;
        }
    }
}

// ---------------- Kernel 2: 4-NN weighted gather -> f16 panel in LDS -> WMMA GEMM + bias ----
__global__ __launch_bounds__(32) void agg_gemm_kernel(const float* __restrict__ x,
                                                      const int*   __restrict__ idx,
                                                      const float* __restrict__ wgt,
                                                      const _Float16* __restrict__ Wh,
                                                      const float* __restrict__ bias,
                                                      float* __restrict__ out) {
    __shared__ _Float16 aggh[16][DIM];           // 16 rows x 128 K, f16 (4 KB)

    const int lane = threadIdx.x;                // wave32, EXEC all ones
    const int row0 = blockIdx.x * 16;            // tiles never straddle b (N%16==0)
    const int b    = row0 / NN;

    const float4* x4 = (const float4*)x;         // [B*N][32] float4

    for (int r = 0; r < 16; ++r) {
        const int   row = row0 + r;
        const int*   ip = idx + row * TOPK;
        const float* wp = wgt + row * TOPK;
        float4 acc = make_float4(0.f, 0.f, 0.f, 0.f);
        #pragma unroll
        for (int k = 0; k < TOPK; ++k) {
            int   j = ip[k];
            float w = wp[k];
            float4 v = x4[((size_t)b * NN + j) * 32 + lane];
            acc.x = fmaf(w, v.x, acc.x);
            acc.y = fmaf(w, v.y, acc.y);
            acc.z = fmaf(w, v.z, acc.z);
            acc.w = fmaf(w, v.w, acc.w);
        }
        v4h h = { (_Float16)acc.x, (_Float16)acc.y, (_Float16)acc.z, (_Float16)acc.w };
        *(v4h*)&aggh[r][lane * 4] = h;           // ds_store_b64
    }
    __syncthreads();

    const int m    = lane & 15;
    const int half = lane >> 4;

    // A fragments (16x32 f16): elems 0..7 at K=kbase+half*8+e, 8..15 at +16
    v16h A[4];
    #pragma unroll
    for (int kc = 0; kc < 4; ++kc) {
        v8h lo = *(const v8h*)&aggh[m][kc * 32 + half * 8];
        v8h hi = *(const v8h*)&aggh[m][kc * 32 + 16 + half * 8];
        #pragma unroll
        for (int e = 0; e < 8; ++e) { A[kc][e] = lo[e]; A[kc][8 + e] = hi[e]; }
    }

    #pragma unroll
    for (int nt = 0; nt < 8; ++nt) {
        v8f c = {};
        #pragma unroll
        for (int kc = 0; kc < 4; ++kc) {
            v16h Bf = *(const v16h*)&Wh[(((nt * 4 + kc) * 32) + lane) * 16];
            c = __builtin_amdgcn_wmma_f32_16x16x32_f16(
                    false, A[kc], false, Bf, (short)0, c, false, false);
        }
        float bv = bias[nt * 16 + m];
        #pragma unroll
        for (int r = 0; r < 8; ++r) {
            int orow = row0 + r + 8 * half;
            out[(size_t)orow * DIM + nt * 16 + m] = c[r] + bv;
        }
    }
}

extern "C" void kernel_launch(void* const* d_in, const int* in_sizes, int n_in,
                              void* d_out, int out_size, void* d_ws, size_t ws_size,
                              hipStream_t stream) {
    const float* x    = (const float*)d_in[0];   // [B,N,128]
    const float* pos  = (const float*)d_in[1];   // [B,N,3]
    const float* W    = (const float*)d_in[2];   // [128,128]
    const float* bias = (const float*)d_in[3];   // [128]
    float* out = (float*)d_out;                  // [B,N,128] f32

    char* ws = (char*)d_ws;
    int*      nidx = (int*)ws;                        // B*N*4 ints   (128 KB)
    float*    nwgt = (float*)(ws + 131072);           // B*N*4 floats (128 KB)
    _Float16* Wh   = (_Float16*)(ws + 262144);        // 16384 halves ( 32 KB)
    float4*   pos4 = (float4*)(ws + 294912);          // B*N float4   (128 KB)

    prep_kernel<<<96, 256, 0, stream>>>(W, pos, Wh, pos4);
    topk_kernel<<<(BB * NN) / 8, 256, 0, stream>>>(pos4, nidx, nwgt);
    agg_gemm_kernel<<<(BB * NN) / 16, 32, 0, stream>>>(x, nidx, nwgt, Wh, bias, out);
}
